// rnn_lut_77180562309394
// MI455X (gfx1250) — compile-verified, matrix operand order checked by reading, not run on hardware
//
#include <hip/hip_runtime.h>
#include <stdint.h>

// Problem constants (match reference)
#define B_N     4096
#define TSEQ    64
#define RNNIN   128
#define HIDDEN  384
#define LABELS  100
#define C1N     64
#define C2N     48
#define K1      (C1N * 16)      // 1024 = one-hot width for step GEMM
#define NT1     (HIDDEN / 16)   // 24 n-tiles of 16

typedef __attribute__((ext_vector_type(16))) __bf16 v16bf;
typedef __attribute__((ext_vector_type(8)))  float  v8f;

static __device__ __forceinline__ unsigned short f32_to_bf16(float f) {
  union { float f; unsigned u; } x; x.f = f;
  unsigned u = x.u;
  return (unsigned short)((u + 0x7FFFu + ((u >> 16) & 1u)) >> 16);
}

// ---------------------------------------------------------------------------
// Kernel 1: embedding precompute  emb[b,t,:] = lenLUT[x0] + ipdLUT[x1]
// One wave per (b,t): 32 lanes x float4 = 128 floats.
// ---------------------------------------------------------------------------
__global__ __launch_bounds__(256) void emb_kernel(
    const int* __restrict__ x, const float* __restrict__ lenLUT,
    const float* __restrict__ ipdLUT, float* __restrict__ emb) {
  int gid  = blockIdx.x * 256 + threadIdx.x;     // B*T*32 threads
  int pair = gid >> 5;                           // b*T + t
  int d4   = gid & 31;
  int i0 = x[pair * 2 + 0];
  int i1 = x[pair * 2 + 1];
  const float4* l4 = (const float4*)lenLUT;
  const float4* p4 = (const float4*)ipdLUT;
  float4 a = l4[(size_t)i0 * 32 + d4];
  float4 b = p4[(size_t)i1 * 32 + d4];
  float4 o; o.x = a.x + b.x; o.y = a.y + b.y; o.z = a.z + b.z; o.w = a.w + b.w;
  ((float4*)emb)[(size_t)pair * 32 + d4] = o;
}

// ---------------------------------------------------------------------------
// Kernel 2: convert LUT1 (1024 x 384 f32, K-major rows) into bf16 stored in
// exact WMMA B-fragment order: [ntile(24)][kblock(32)][lane(32)][pos(16)].
// B-frag mapping (32x16 K x N, 16-bit): lane = (n%16) + 16*((kk/8)&1),
// pos = 8*(kk>=16) + kk%8.  Each lane's fragment = 32 contiguous bytes.
// ---------------------------------------------------------------------------
__global__ __launch_bounds__(256) void lut1_frag_kernel(
    const float* __restrict__ lut1, unsigned short* __restrict__ frag) {
  int gid = blockIdx.x * 256 + threadIdx.x;      // K1*HIDDEN elements
  int k = gid / HIDDEN;
  int n = gid % HIDDEN;
  int kb = k >> 5, kk = k & 31;
  int lane = (n & 15) + (((kk >> 3) & 1) << 4);
  int pos  = ((kk >> 4) << 3) + (kk & 7);
  int nt   = n >> 4;
  frag[((size_t)(nt * 32 + kb) * 32 + lane) * 16 + pos] = f32_to_bf16(lut1[gid]);
}

// ---------------------------------------------------------------------------
// Kernel 3 (per step, FUSED): front-end + one-hot GEMM.
//
// Phase A (all 256 threads): for the block's 64 rows x 64 chunks:
//   z = rnnin_chunk(8) @ S1[c] - T1[c]; s = sign(z); y = s @ H1;
//   sIdx[r][c] = argmax(y)          (rnnin = [emb(:,t,:) | h_in])
//
// Phase B (8 waves): h_new = OneHot(idx) (64 x 1024) @ LUT1 (1024 x 384)
// via v_wmma_f32_16x16x32_bf16. A-fragments synthesized in registers from
// the 4-bit codes (2 ones per 32-wide K block, one per chunk):
//   chunk c0=2kb: active if ((t0>>3)&1)==laneHi, pos = t0&7
//   chunk c1=2kb+1: active if ((t1>>3)&1)==laneHi, pos = 8 + (t1&7)
// Wave owns 1 ntile (blockIdx.y*8 + wave), 4 m-tiles of accumulators,
// 32 K-blocks => 128 WMMAs/wave.  B fragment for kb+1 is prefetched while
// the 4 WMMAs of kb execute.
// ---------------------------------------------------------------------------
__global__ __launch_bounds__(256) void step_kernel(
    const float* __restrict__ emb, const float* __restrict__ h_in,
    const float* __restrict__ S1, const float* __restrict__ T1,
    const float* __restrict__ H1,
    const unsigned short* __restrict__ lut_frag,
    float* __restrict__ h_out, int t, int first) {
  __shared__ float sS[C1N * 8 * 15];             // 30 KB
  __shared__ float sT[C1N * 15];
  __shared__ float sH[15 * 16];
  __shared__ unsigned char sIdx[64 * 64];        // 4 KB code tile
  int tid = threadIdx.x;
  int rowBase = blockIdx.x * 64;

  for (int i = tid; i < C1N * 8 * 15; i += 256) sS[i] = S1[i];
  for (int i = tid; i < C1N * 15;     i += 256) sT[i] = T1[i];
  for (int i = tid; i < 240;          i += 256) sH[i] = H1[i];
  __syncthreads();

  // ---- Phase A: codes for 64 rows x 64 chunks (16 pairs per thread) ----
  for (int it = 0; it < 16; ++it) {
    int p = it * 256 + tid;                 // 0..4095
    int rl = p >> 6;                        // local row 0..63
    int c  = p & 63;
    int r  = rowBase + rl;
    float v[8];
    if (c < 16) {
      const float* e = emb + ((size_t)r * TSEQ + t) * RNNIN + c * 8;
      #pragma unroll
      for (int j = 0; j < 8; ++j) v[j] = e[j];
    } else if (first) {
      #pragma unroll
      for (int j = 0; j < 8; ++j) v[j] = 0.f;
    } else {
      const float* hp = h_in + (size_t)r * HIDDEN + (c - 16) * 8;
      #pragma unroll
      for (int j = 0; j < 8; ++j) v[j] = hp[j];
    }
    float z[15];
    #pragma unroll
    for (int k = 0; k < 15; ++k) z[k] = -sT[c * 15 + k];
    #pragma unroll
    for (int j = 0; j < 8; ++j) {
      float vj = v[j];
      #pragma unroll
      for (int k = 0; k < 15; ++k) z[k] += vj * sS[(c * 8 + j) * 15 + k];
    }
    #pragma unroll
    for (int k = 0; k < 15; ++k) z[k] = (z[k] > 0.f) ? 1.f : -1.f;
    float best = -1e30f; int bi = 0;
    #pragma unroll
    for (int jj = 0; jj < 16; ++jj) {
      float y = 0.f;
      #pragma unroll
      for (int k = 0; k < 15; ++k) y += z[k] * sH[k * 16 + jj];
      if (y > best) { best = y; bi = jj; }
    }
    sIdx[rl * 64 + c] = (unsigned char)bi;
  }
  __syncthreads();

  // ---- Phase B: WMMA GEMM against fragment-ordered bf16 LUT1 ----
  int wave = tid >> 5;
  int lane = tid & 31;
  unsigned hi = (unsigned)(lane >> 4);
  int m = lane & 15;
  int nt = blockIdx.y * 8 + wave;                // 0..23

  v8f acc[4];
  #pragma unroll
  for (int mt = 0; mt < 4; ++mt)
    #pragma unroll
    for (int q = 0; q < 8; ++q) acc[mt][q] = 0.f;

  const v16bf* bptr = (const v16bf*)lut_frag + (size_t)nt * 32 * 32 + lane;
  v16bf bfrag = bptr[0];

  for (int kb = 0; kb < 32; ++kb) {
    v16bf bnext = bptr[((kb + 1) & 31) * 32];    // prefetch next K-block
    #pragma unroll
    for (int mt = 0; mt < 4; ++mt) {
      int r = mt * 16 + m;
      unsigned t0 = sIdx[r * 64 + 2 * kb];
      unsigned t1 = sIdx[r * 64 + 2 * kb + 1];
      unsigned e0 = t0 & 7u, e1 = t1 & 7u;
      unsigned p0 = (((t0 >> 3) & 1u) == hi) ? (0x3F80u << ((e0 & 1u) * 16)) : 0u;
      unsigned p1 = (((t1 >> 3) & 1u) == hi) ? (0x3F80u << ((e1 & 1u) * 16)) : 0u;
      unsigned w0 = e0 >> 1, w1 = e1 >> 1;
      union { unsigned u[8]; v16bf v; } af;
      #pragma unroll
      for (int i = 0; i < 4; ++i) af.u[i]     = (w0 == (unsigned)i) ? p0 : 0u;
      #pragma unroll
      for (int i = 0; i < 4; ++i) af.u[4 + i] = (w1 == (unsigned)i) ? p1 : 0u;
      acc[mt] = __builtin_amdgcn_wmma_f32_16x16x32_bf16(
          false, af.v, false, bfrag, (short)0, acc[mt], false, false);
    }
    bfrag = bnext;
  }

  // C/D layout: vgpr q -> row q + 8*hi, col = lane%16
  #pragma unroll
  for (int mt = 0; mt < 4; ++mt) {
    #pragma unroll
    for (int q = 0; q < 8; ++q) {
      int row = rowBase + mt * 16 + q + 8 * (int)hi;
      int col = nt * 16 + m;
      h_out[(size_t)row * HIDDEN + col] = acc[mt][q];
    }
  }
}

// ---------------------------------------------------------------------------
// Kernel 4: final decode (C2=48 chunks -> 100 labels) + log_softmax.
// One wave per batch row; lanes cooperate on the 100-wide accumulator and
// reduce max/sum via shuffles. Runs once -> VALU path is fine.
// ---------------------------------------------------------------------------
__global__ __launch_bounds__(256) void final_kernel(
    const float* __restrict__ h_last, const float* __restrict__ S2,
    const float* __restrict__ T2, const float* __restrict__ H2,
    const float* __restrict__ LUT2, float* __restrict__ out) {
  __shared__ float sS[C2N * 8 * 15];
  __shared__ float sT[C2N * 15];
  __shared__ float sH[15 * 16];
  __shared__ unsigned char sIdx[8 * C2N];
  int tid = threadIdx.x;
  for (int i = tid; i < C2N * 8 * 15; i += 256) sS[i] = S2[i];
  for (int i = tid; i < C2N * 15;     i += 256) sT[i] = T2[i];
  for (int i = tid; i < 240;          i += 256) sH[i] = H2[i];
  __syncthreads();

  int wave = tid >> 5, lane = tid & 31;
  int row = blockIdx.x * 8 + wave;
  const float* hr = h_last + (size_t)row * HIDDEN;

  #pragma unroll
  for (int pass = 0; pass < 2; ++pass) {
    int c = lane + pass * 32;
    if (c < C2N) {
      float z[15];
      #pragma unroll
      for (int k = 0; k < 15; ++k) z[k] = -sT[c * 15 + k];
      #pragma unroll
      for (int j = 0; j < 8; ++j) {
        float vj = hr[c * 8 + j];
        #pragma unroll
        for (int k = 0; k < 15; ++k) z[k] += vj * sS[(c * 8 + j) * 15 + k];
      }
      #pragma unroll
      for (int k = 0; k < 15; ++k) z[k] = (z[k] > 0.f) ? 1.f : -1.f;
      float best = -1e30f; int bi = 0;
      #pragma unroll
      for (int jj = 0; jj < 16; ++jj) {
        float y = 0.f;
        #pragma unroll
        for (int k = 0; k < 15; ++k) y += z[k] * sH[k * 16 + jj];
        if (y > best) { best = y; bi = jj; }
      }
      sIdx[wave * C2N + c] = (unsigned char)bi;
    }
  }
  __syncthreads();

  float a0 = 0.f, a1 = 0.f, a2 = 0.f, a3 = 0.f;
  for (int c = 0; c < C2N; ++c) {
    int tt = sIdx[wave * C2N + c];
    const float* L = LUT2 + (size_t)(c * 16 + tt) * LABELS;
    a0 += L[lane]; a1 += L[lane + 32]; a2 += L[lane + 64];
    if (lane < 4) a3 += L[lane + 96];
  }
  float lm = fmaxf(fmaxf(a0, a1), a2);
  if (lane < 4) lm = fmaxf(lm, a3);
  for (int off = 16; off >= 1; off >>= 1) lm = fmaxf(lm, __shfl_xor(lm, off));
  float se = expf(a0 - lm) + expf(a1 - lm) + expf(a2 - lm) +
             ((lane < 4) ? expf(a3 - lm) : 0.f);
  for (int off = 16; off >= 1; off >>= 1) se += __shfl_xor(se, off);
  float ls = logf(se);
  out[(size_t)row * LABELS + lane]      = a0 - lm - ls;
  out[(size_t)row * LABELS + lane + 32] = a1 - lm - ls;
  out[(size_t)row * LABELS + lane + 64] = a2 - lm - ls;
  if (lane < 4) out[(size_t)row * LABELS + lane + 96] = a3 - lm - ls;
}

// ---------------------------------------------------------------------------
// Workspace layout (bytes):
//   emb   f32 : [0, 134217728)             B*T*128*4
//   hA    f32 : [134217728, +6291456)      B*384*4
//   hB    f32 : [140509184, +6291456)
//   lfrag bf16: [146800640, +786432)       24*32*32*16*2
// Total ~140 MB.
// ---------------------------------------------------------------------------
extern "C" void kernel_launch(void* const* d_in, const int* in_sizes, int n_in,
                              void* d_out, int out_size, void* d_ws, size_t ws_size,
                              hipStream_t stream) {
  const int*   x      = (const int*)d_in[0];
  const float* lenLUT = (const float*)d_in[1];
  const float* ipdLUT = (const float*)d_in[2];
  const float* S1     = (const float*)d_in[3];
  const float* H1     = (const float*)d_in[4];
  const float* T1     = (const float*)d_in[5];
  const float* LUT1   = (const float*)d_in[6];
  const float* S2     = (const float*)d_in[7];
  const float* H2     = (const float*)d_in[8];
  const float* T2     = (const float*)d_in[9];
  const float* LUT2   = (const float*)d_in[10];
  float* out = (float*)d_out;

  char* ws = (char*)d_ws;
  float*          emb   = (float*)(ws);
  float*          hA    = (float*)(ws + 134217728);
  float*          hB    = (float*)(ws + 134217728 + 6291456);
  unsigned short* lfrag = (unsigned short*)(ws + 134217728 + 2 * 6291456);

  emb_kernel<<<(B_N * TSEQ * 32) / 256, 256, 0, stream>>>(x, lenLUT, ipdLUT, emb);
  lut1_frag_kernel<<<(K1 * HIDDEN) / 256, 256, 0, stream>>>(LUT1, lfrag);

  float* h_out = hA;
  for (int t = 0; t < TSEQ; ++t) {
    h_out = (t & 1) ? hB : hA;
    const float* h_in = (t & 1) ? hA : hB;
    step_kernel<<<dim3(B_N / 64, 3), 256, 0, stream>>>(
        emb, h_in, S1, T1, H1, lfrag, h_out, t, (t == 0) ? 1 : 0);
  }
  final_kernel<<<B_N / 8, 256, 0, stream>>>(h_out, S2, T2, H2, LUT2, out);
}
